// QuantizedQwen3MoeSparseMoeBlock_39865886442067
// MI455X (gfx1250) — matrix operand order repaired and checked by reading, not compiled, not run on hardware
//
#include <hip/hip_runtime.h>
#include <hip/hip_bf16.h>

#define HDIM 2048
#define NEXP 16
#define IDIM 768
#define TTOK 2048          // B*S = 2*1024
#define NPAIR (TTOK * 2)   // top-2 routing -> 4096 token-expert pairs

typedef __bf16 bf16_t;
typedef __attribute__((ext_vector_type(16))) __bf16 v16bf;
typedef __attribute__((ext_vector_type(8)))  __bf16 v8bf_t;
typedef __attribute__((ext_vector_type(4)))  __bf16 v4bf_t;
typedef __attribute__((ext_vector_type(8)))  float  v8f;
typedef __attribute__((ext_vector_type(4)))  float  v4f;

__device__ __forceinline__ v8f v8f_zero() {
  v8f z;
#pragma unroll
  for (int i = 0; i < 8; ++i) z[i] = 0.f;
  return z;
}

// fast reciprocal (v_rcp_f32), ~1 ulp: plenty for a bf16-bound activation
__device__ __forceinline__ float fast_rcp(float x) {
  return __builtin_amdgcn_rcpf(x);
}

// A-fragment (16x32 bf16), contiguous K order matching B:
// lanes 0-15: row M=lane,    K=0..15
// lanes16-31: row M=lane-16, K=16..31
__device__ __forceinline__ v16bf load_a_bf16(const bf16_t* __restrict__ rowp, int k0) {
  const bf16_t* p = rowp + k0;
  v8bf_t c0 = *(const v8bf_t*)(p);
  v8bf_t c1 = *(const v8bf_t*)(p + 8);
  v16bf a;
#pragma unroll
  for (int i = 0; i < 8; ++i) { a[i] = c0[i]; a[8 + i] = c1[i]; }
  return a;
}

// B-fragment (32x16 bf16) in contiguous K order (same as A), built from fp32 row-major
// weights W[n][k] (k contiguous): lane<16 -> col N=lane, K=0..15;
// lane>=16 -> col N=lane-16, K=16..31.
__device__ __forceinline__ v16bf load_b_f32(const float* __restrict__ lanep, int k0) {
  const float* p = lanep + k0;
  v16bf b;
#pragma unroll
  for (int i = 0; i < 4; ++i) {
    v4f f = *(const v4f*)(p + i * 4);
    b[i * 4 + 0] = (bf16_t)f[0];
    b[i * 4 + 1] = (bf16_t)f[1];
    b[i * 4 + 2] = (bf16_t)f[2];
    b[i * 4 + 3] = (bf16_t)f[3];
  }
  return b;
}

// ---------------- Router: logits, top-2, per-expert slot assignment ----------
__global__ void __launch_bounds__(256) moe_router(
    const float* __restrict__ x, const float* __restrict__ gw,
    float* __restrict__ logits_out, int* __restrict__ cnt,
    int* __restrict__ pair_e, int* __restrict__ pair_slot,
    float* __restrict__ pair_w) {
  const int lane = threadIdx.x & 31;
  const int wl   = threadIdx.x >> 5;
  const int t    = blockIdx.x * 8 + wl;
  __shared__ float lsm[8][NEXP];
  const float* xr = x + (size_t)t * HDIM;
  for (int e = 0; e < NEXP; ++e) {
    const float* wr = gw + (size_t)e * HDIM;
    float acc = 0.f;
#pragma unroll 8
    for (int j = 0; j < HDIM / 32; ++j) acc += xr[j * 32 + lane] * wr[j * 32 + lane];
#pragma unroll
    for (int off = 16; off > 0; off >>= 1) acc += __shfl_down(acc, off, 32);
    if (lane == 0) lsm[wl][e] = acc;
  }
  if (lane < NEXP) logits_out[(size_t)t * NEXP + lane] = lsm[wl][lane];
  if (lane == 0) {
    float l[NEXP];
    for (int e = 0; e < NEXP; ++e) l[e] = lsm[wl][e];
    int i0 = 0;
    for (int e = 1; e < NEXP; ++e) if (l[e] > l[i0]) i0 = e;
    int i1 = (i0 == 0) ? 1 : 0;
    for (int e = 0; e < NEXP; ++e) if (e != i0 && l[e] > l[i1]) i1 = e;
    // normalized top-2 softmax weights: w0 = e^l0/(e^l0+e^l1); l1<=l0 -> stable
    float w0 = 1.f / (1.f + __expf(l[i1] - l[i0]));
    float w1 = 1.f - w0;
    const int p = t * 2;
    int s0 = atomicAdd(&cnt[i0], 1);
    int s1 = atomicAdd(&cnt[i1], 1);
    pair_e[p] = i0; pair_slot[p] = s0; pair_w[p] = w0;
    pair_e[p + 1] = i1; pair_slot[p + 1] = s1; pair_w[p + 1] = w1;
  }
}

__global__ void moe_prefix(const int* __restrict__ cnt, int* __restrict__ basep) {
  if (threadIdx.x == 0) {
    int a = 0;
    for (int e = 0; e < NEXP; ++e) { basep[e] = a; a += cnt[e]; }
    basep[NEXP] = a;
  }
}

// ------------- Gather: compacted, expert-grouped bf16 token rows -------------
__global__ void __launch_bounds__(256) moe_gather(
    const float* __restrict__ x, const int* __restrict__ pair_e,
    const int* __restrict__ pair_slot, const int* __restrict__ basep,
    int* __restrict__ cidx, bf16_t* __restrict__ Xg) {
  const int lane = threadIdx.x & 31;
  const int p = blockIdx.x * 8 + (threadIdx.x >> 5);
  const int e = pair_e[p];
  const int dst = basep[e] + pair_slot[p];
  if (lane == 0) cidx[p] = dst;
  const float* src = x + (size_t)(p >> 1) * HDIM;
  bf16_t* d = Xg + (size_t)dst * HDIM;
#pragma unroll
  for (int j = 0; j < HDIM / 128; ++j) {
    v4f f = *(const v4f*)(src + j * 128 + lane * 4);
    v4bf_t o;
    o[0] = (bf16_t)f[0]; o[1] = (bf16_t)f[1];
    o[2] = (bf16_t)f[2]; o[3] = (bf16_t)f[3];
    *(v4bf_t*)(d + j * 128 + lane * 4) = o;
  }
}

// -------- gate_up GEMM + SwiGLU: per wave 16 cols x 64 rows, K step 32 -------
__global__ void __launch_bounds__(256) moe_gate_up(
    const float* __restrict__ gate_up_w, const bf16_t* __restrict__ Xg,
    const int* __restrict__ cnt, const int* __restrict__ basep,
    bf16_t* __restrict__ Hb) {
  const int lane = threadIdx.x & 31;
  const int wv   = threadIdx.x >> 5;
  const int e    = blockIdx.y;
  const int nw   = blockIdx.x * 128 + wv * 16;  // h-column base of this wave
  const int Ne = cnt[e];
  const int rowStart = basep[e];
  const float* W1 = gate_up_w + (size_t)e * (2 * IDIM) * HDIM;
  const int hi = (lane >> 4) & 1;
  const int m  = lane & 15;
  // per-lane weight row pointers (B layout: col = lane&15, K half by lane>>4)
  const float* Wg = W1 + (size_t)(nw + m) * HDIM + hi * 16;
  const float* Wu = W1 + (size_t)(IDIM + nw + m) * HDIM + hi * 16;
  const int aoff = hi * 16;  // A layout K sub-chunk offset

  for (int m0 = blockIdx.z * 64; m0 < Ne; m0 += 64 * gridDim.z) {
    v8f accg[4], accu[4];
    const bf16_t* arow[4];
#pragma unroll
    for (int i = 0; i < 4; ++i) {
      accg[i] = v8f_zero();
      accu[i] = v8f_zero();
      int row = rowStart + m0 + i * 16 + m;
      row = row < (NPAIR - 1) ? row : (NPAIR - 1);
      arow[i] = Xg + (size_t)row * HDIM + aoff;
    }
    for (int k0 = 0; k0 < HDIM; k0 += 32) {
      v16bf bg = load_b_f32(Wg, k0);
      v16bf bu = load_b_f32(Wu, k0);
#pragma unroll
      for (int i = 0; i < 4; ++i) {
        v16bf a = load_a_bf16(arow[i], k0);
        accg[i] = __builtin_amdgcn_wmma_f32_16x16x32_bf16(
            false, a, false, bg, (short)0, accg[i], false, false);
        accu[i] = __builtin_amdgcn_wmma_f32_16x16x32_bf16(
            false, a, false, bu, (short)0, accu[i], false, false);
      }
    }
    const int col = nw + m;
#pragma unroll
    for (int i = 0; i < 4; ++i) {
#pragma unroll
      for (int r = 0; r < 8; ++r) {
        int lrow = m0 + i * 16 + hi * 8 + r;
        if (lrow < Ne) {
          float g = accg[i][r], u = accu[i][r];
          // silu(g)*u with fast rcp (v_rcp_f32): bf16-accurate
          float h = g * fast_rcp(1.f + __expf(-g)) * u;
          Hb[(size_t)(rowStart + lrow) * IDIM + col] = (bf16_t)h;
        }
      }
    }
  }
}

// -------------- down GEMM: per wave 16 H-cols x 64 rows, K over I ------------
__global__ void __launch_bounds__(256) moe_down(
    const float* __restrict__ down_w, const bf16_t* __restrict__ Hb,
    const int* __restrict__ cnt, const int* __restrict__ basep,
    float* __restrict__ Y) {
  const int lane = threadIdx.x & 31;
  const int wv   = threadIdx.x >> 5;
  const int e    = blockIdx.y;
  const int nw   = blockIdx.x * 128 + wv * 16;  // output H-column base
  const int Ne = cnt[e];
  const int rowStart = basep[e];
  const int hi = (lane >> 4) & 1;
  const int m  = lane & 15;
  const float* W2 = down_w + (size_t)e * HDIM * IDIM + (size_t)(nw + m) * IDIM + hi * 16;
  const int aoff = hi * 16;

  for (int m0 = blockIdx.z * 64; m0 < Ne; m0 += 64 * gridDim.z) {
    v8f acc[4];
    const bf16_t* arow[4];
#pragma unroll
    for (int i = 0; i < 4; ++i) {
      acc[i] = v8f_zero();
      int row = rowStart + m0 + i * 16 + m;
      row = row < (NPAIR - 1) ? row : (NPAIR - 1);
      arow[i] = Hb + (size_t)row * IDIM + aoff;
    }
    for (int k0 = 0; k0 < IDIM; k0 += 32) {
      v16bf b = load_b_f32(W2, k0);
#pragma unroll
      for (int i = 0; i < 4; ++i) {
        v16bf a = load_a_bf16(arow[i], k0);
        acc[i] = __builtin_amdgcn_wmma_f32_16x16x32_bf16(
            false, a, false, b, (short)0, acc[i], false, false);
      }
    }
    const int col = nw + m;
#pragma unroll
    for (int i = 0; i < 4; ++i) {
#pragma unroll
      for (int r = 0; r < 8; ++r) {
        int lrow = m0 + i * 16 + hi * 8 + r;
        if (lrow < Ne) Y[(size_t)(rowStart + lrow) * HDIM + col] = acc[i][r];
      }
    }
  }
}

// ------------- Combine: out[t] = w0*Y[c0] + w1*Y[c1] (fixed order) -----------
__global__ void __launch_bounds__(256) moe_reduce(
    const float* __restrict__ Y, const int* __restrict__ cidx,
    const float* __restrict__ pair_w, float* __restrict__ out) {
  const int t = blockIdx.x;
  const float w0 = pair_w[t * 2], w1 = pair_w[t * 2 + 1];
  const float* y0 = Y + (size_t)cidx[t * 2] * HDIM;
  const float* y1 = Y + (size_t)cidx[t * 2 + 1] * HDIM;
  float* o = out + (size_t)t * HDIM;
#pragma unroll
  for (int j = 0; j < 2; ++j) {
    int idx = j * 1024 + threadIdx.x * 4;
    v4f a = *(const v4f*)(y0 + idx);
    v4f b = *(const v4f*)(y1 + idx);
    v4f r;
#pragma unroll
    for (int c = 0; c < 4; ++c) r[c] = w0 * a[c] + w1 * b[c];
    *(v4f*)(o + idx) = r;
  }
}

extern "C" void kernel_launch(void* const* d_in, const int* in_sizes, int n_in,
                              void* d_out, int out_size, void* d_ws, size_t ws_size,
                              hipStream_t stream) {
  (void)in_sizes; (void)n_in; (void)out_size; (void)ws_size;
  const float* x   = (const float*)d_in[0];  // [T, H]
  const float* gw  = (const float*)d_in[1];  // [E, H]
  const float* guw = (const float*)d_in[2];  // [E, 2I, H]
  const float* dww = (const float*)d_in[3];  // [E, H, I]
  float* out = (float*)d_out;                          // [T, H]
  float* logits_out = out + (size_t)TTOK * HDIM;       // [T, E]

  char* ws = (char*)d_ws;
  int*   cnt    = (int*)(ws + 0);                      // 16 ints (256B region)
  int*   basep  = (int*)(ws + 256);                    // 17 ints
  int*   pair_e = (int*)(ws + 512);                    // NPAIR ints
  int*   pair_s = (int*)(ws + 512 + 16384);
  float* pair_w = (float*)(ws + 512 + 2 * 16384);
  int*   cidx   = (int*)(ws + 512 + 3 * 16384);
  size_t off = 512 + 4 * 16384;                        // 66048 (256B aligned)
  bf16_t* Xg = (bf16_t*)(ws + off); off += (size_t)NPAIR * HDIM * 2;  // 16 MB
  bf16_t* Hb = (bf16_t*)(ws + off); off += (size_t)NPAIR * IDIM * 2;  //  6 MB
  float*  Y  = (float*)(ws + off);                     // NPAIR*H fp32, 33.5 MB

  hipMemsetAsync(cnt, 0, 256, stream);
  moe_router<<<TTOK / 8, 256, 0, stream>>>(x, gw, logits_out, cnt, pair_e, pair_s, pair_w);
  moe_prefix<<<1, 32, 0, stream>>>(cnt, basep);
  moe_gather<<<NPAIR / 8, 256, 0, stream>>>(x, pair_e, pair_s, basep, cidx, Xg);
  moe_gate_up<<<dim3(6, 16, 4), 256, 0, stream>>>(guw, Xg, cnt, basep, Hb);
  moe_down<<<dim3(16, 16, 4), 256, 0, stream>>>(dww, Hb, cnt, basep, Y);
  moe_reduce<<<TTOK, 256, 0, stream>>>(Y, cidx, pair_w, out);
}
